// Encoding3D_30193620091340
// MI455X (gfx1250) — matrix operand order repaired and checked by reading, not compiled, not run on hardware
//
#include <hip/hip_runtime.h>

typedef __attribute__((ext_vector_type(16))) _Float16 v16h;
typedef __attribute__((ext_vector_type(8)))  float    v8f;

#define LOG2E 1.4426950408889634f
// B=4, D=64, K=32, N = 8*32*32 = 8192

// ---------------- K0: zero the E_glob accumulator (256 floats) ----------------
__global__ void enc3d_zero(float* __restrict__ eglob) {
    eglob[threadIdx.x] = 0.0f;
}

// ---------------- K1: softmax-encoding over K, write E, accumulate E_glob ----
// grid: B*D*(N/256) blocks of 256 threads; each block handles one (b,d) and a
// 256-wide n chunk, so codewords[k*64+d] / scale[k*64+d] are wave-uniform
// (scalar-cache broadcasts), and X / E accesses are perfectly coalesced.
__global__ void enc3d_encode(const float* __restrict__ X,
                             const float* __restrict__ cw,
                             const float* __restrict__ sc,
                             float* __restrict__ E,       // (B, D, N) == d_out
                             float* __restrict__ eglob) { // (B, D) sum over n
    const int chunk = blockIdx.x & 31;           // 32 chunks of 256 over N=8192
    const int d     = (blockIdx.x >> 5) & 63;
    const int b     = blockIdx.x >> 11;
    const int n     = (chunk << 8) + threadIdx.x;
    const int base  = ((b << 6) + d) << 13;      // (b*64+d)*8192

    const float x = X[base + n];

    float sum_e = 0.0f;
    float num   = 0.0f;
#pragma unroll
    for (int k = 0; k < 32; ++k) {
        const float c  = cw[(k << 6) + d];       // uniform -> s_load
        const float s2 = sc[(k << 6) + d] * LOG2E;
        const float r  = x - c;
        const float e  = __builtin_amdgcn_exp2f(s2 * (r * r)); // v_exp_f32
        sum_e += e;
        num = fmaf(e, r, num);
    }
    // softmax-weighted mean: num / sum_e via raw v_rcp_f32 (sum_e >= ~1e-11,
    // args are <= 0 so sum is in (0, 32]; ~1 ulp rcp is plenty here)
    const float ev = num * __builtin_amdgcn_rcpf(sum_e);
    E[base + n] = ev;

    // block-level sum over the 256 n's, one atomic per block
    __shared__ float red[256];
    red[threadIdx.x] = ev;
    __syncthreads();
#pragma unroll
    for (int off = 128; off > 0; off >>= 1) {
        if (threadIdx.x < (unsigned)off) red[threadIdx.x] += red[threadIdx.x + off];
        __syncthreads();
    }
    if (threadIdx.x == 0) atomicAdd(&eglob[(b << 6) + d], red[0]);
}

// ---------------- K2: gamma = sigmoid(E_glob/K @ fc_w^T + fc_b) via WMMA -----
// One wave32. M=4 rows padded to 16, K=64 (2 k-splits of 32), N=64 (4 tiles).
// Emits 8x v_wmma_f32_16x16x32_f16.
__global__ void enc3d_gamma(const float* __restrict__ eglob,
                            const float* __restrict__ fc_w, // (D_out, D_in) row-major
                            const float* __restrict__ fc_b,
                            float* __restrict__ gamma) {
    const int lane = threadIdx.x;                // 0..31, EXEC all ones
    const int m    = lane & 15;                  // A-matrix row
    const int kb   = (lane < 16) ? 0 : 16;       // A-matrix K base per half-wave
    const float invK = 1.0f / 32.0f;             // mean over K codewords

    // A fragments: contiguous k order, same as B — lane holds M=m, halves h:
    //   h in [0,16) -> K = kb + h
    v16h a0, a1;
#pragma unroll
    for (int h = 0; h < 16; ++h) {
        const int kk = kb + h;
        float v0 = eglob[((m & 3) << 6) + kk]       * invK;
        float v1 = eglob[((m & 3) << 6) + 32 + kk]  * invK;
        if (m >= 4) { v0 = 0.0f; v1 = 0.0f; }      // pad rows 4..15 with zeros
        a0[h] = (_Float16)v0;
        a1[h] = (_Float16)v1;
    }

    const int nl = lane & 15;                    // B-matrix column (N)
    const int kg = (lane < 16) ? 0 : 16;         // B-matrix K group per half-wave

#pragma unroll
    for (int t = 0; t < 4; ++t) {                // 4 N-tiles of 16 -> D_out = 64
        v8f c = {};
        v16h b0, b1;
#pragma unroll
        for (int h = 0; h < 16; ++h) {
            // B[k][n] = fc_w[(t*16+n)*64 + k_global]; logits = Eglob x fc_w^T
            b0[h] = (_Float16)fc_w[((t << 4) + nl) * 64 + (kg + h)];
            b1[h] = (_Float16)fc_w[((t << 4) + nl) * 64 + 32 + (kg + h)];
        }
        c = __builtin_amdgcn_wmma_f32_16x16x32_f16(false, a0, false, b0,
                                                   (short)0, c, false, false);
        c = __builtin_amdgcn_wmma_f32_16x16x32_f16(false, a1, false, b1,
                                                   (short)0, c, false, false);
        // D layout: VGPR r, lanes 0..15 -> M=r, N=lane. Rows 0..3 are b=0..3.
        if (lane < 16) {
#pragma unroll
            for (int r = 0; r < 4; ++r) {
                const float logit = c[r] + fc_b[(t << 4) + lane];
                // sigmoid via v_exp_f32 + v_rcp_f32 (no IEEE div chain)
                const float e = __builtin_amdgcn_exp2f(-logit * LOG2E);
                gamma[(r << 6) + (t << 4) + lane] = __builtin_amdgcn_rcpf(1.0f + e);
            }
        }
    }
}

// ---------------- K3: out = relu(E * (1 + gamma)) in place -------------------
__global__ void enc3d_apply(float* __restrict__ out,
                            const float* __restrict__ gamma) {
    const int idx = (blockIdx.x << 8) + threadIdx.x; // B*D*N = 2097152 total
    const int bd  = idx >> 13;                       // (b*64+d)
    const float g = gamma[bd];
    const float e = out[idx];
    const float v = e * (1.0f + g);
    out[idx] = v > 0.0f ? v : 0.0f;
}

extern "C" void kernel_launch(void* const* d_in, const int* in_sizes, int n_in,
                              void* d_out, int out_size, void* d_ws, size_t ws_size,
                              hipStream_t stream) {
    const float* X    = (const float*)d_in[0];  // (4, 64, 8, 32, 32)
    const float* cw   = (const float*)d_in[1];  // (32, 64)
    const float* sc   = (const float*)d_in[2];  // (32, 64)
    const float* fc_w = (const float*)d_in[3];  // (64, 64)
    const float* fc_b = (const float*)d_in[4];  // (64,)
    float* out   = (float*)d_out;               // (4, 64, 8192) flat
    float* eglob = (float*)d_ws;                // 256 floats
    float* gamma = eglob + 256;                 // 256 floats

    enc3d_zero  <<<1, 256, 0, stream>>>(eglob);
    enc3d_encode<<<4 * 64 * 32, 256, 0, stream>>>(X, cw, sc, out, eglob);
    enc3d_gamma <<<1, 32, 0, stream>>>(eglob, fc_w, fc_b, gamma);
    enc3d_apply <<<4 * 64 * 8192 / 256, 256, 0, stream>>>(out, gamma);
}
